// SparseMultiHeadAttention_54254026883873
// MI455X (gfx1250) — matrix operand order repaired and checked by reading, not compiled, not run on hardware
//
#include <hip/hip_runtime.h>
#include <hip/hip_bf16.h>

// Problem constants (B, L, C, H) = (4, 2048, 1024, 16), D = 64
#define BB 4
#define LL 2048
#define CC 1024
#define HH 16
#define DD 64

typedef __attribute__((ext_vector_type(16))) __bf16 v16bf;
typedef __attribute__((ext_vector_type(8)))  float v8f;

__device__ __forceinline__ unsigned short f2bf(float f) {
  union { float f; unsigned u; } a; a.f = f;
  unsigned r = a.u + 0x7fffu + ((a.u >> 16) & 1u);  // round-to-nearest-even
  return (unsigned short)(r >> 16);
}
__device__ __forceinline__ float bf2f(unsigned x) {
  union { unsigned u; float f; } a; a.u = x << 16; return a.f;
}
__device__ __forceinline__ v16bf frag_ld(const unsigned short* p) {
  v16bf r; __builtin_memcpy(&r, p, 32); return r;
}
// A-slot for K (contiguous order, matching B):
// lane-half0 holds K {0..15}, lane-half1 holds K {16..31}.
__device__ __forceinline__ int slotA(int k) {
  return k;
}
__device__ __forceinline__ v8f wmma_bf16(v16bf a, v16bf b, v8f c) {
  return __builtin_amdgcn_wmma_f32_16x16x32_bf16(false, a, false, b, (short)0, c,
                                                 false, false);
}

// ---------------------------------------------------------------------------
// GEMM: C = A(MxK) * W(KxN) + bias. bf16 WMMA, f32 accumulate.
// Block = 256 threads = 8 waves (4 in M, 2 in N); 128x128 tile; BK = 32.
// MODE 0: A is f32 (x), epilogue scatters bf16 into q/k/v buffers.
// MODE 1: A is bf16 (attention output), epilogue stores f32.
// ---------------------------------------------------------------------------
template <int MODE>
__global__ __launch_bounds__(256) void gemm_bf16_wmma(
    const float* __restrict__ Af, const unsigned short* __restrict__ Ab,
    const float* __restrict__ W, const float* __restrict__ bias,
    unsigned short* __restrict__ oq, unsigned short* __restrict__ ok,
    unsigned short* __restrict__ ov, float* __restrict__ of,
    int M, int N, int K) {
  __shared__ __align__(16) unsigned short As[128][32];  // [m][A-slot]
  __shared__ __align__(16) unsigned short Bs[128][32];  // [n][k]
  const int t = threadIdx.x;
  const int wid = t >> 5, lane = t & 31;
  const int wm = wid & 3, wn = wid >> 2;
  const int lh = lane >> 4, ll = lane & 15;
  const int m0 = blockIdx.y * 128, n0 = blockIdx.x * 128;

  v8f acc[2][4];
  const v8f vz = {0.f, 0.f, 0.f, 0.f, 0.f, 0.f, 0.f, 0.f};
#pragma unroll
  for (int i = 0; i < 2; ++i)
#pragma unroll
    for (int j = 0; j < 4; ++j) acc[i][j] = vz;

  const int arow = t >> 3, ak = (t & 7) * 4;   // A stage: 4 rows x float4
  const int bk = t >> 3, bn0 = (t & 7) * 16;   // B stage: 1 k-row x 16 n

  for (int k0 = 0; k0 < K; k0 += 32) {
    if (MODE == 0) {
#pragma unroll
      for (int rr = 0; rr < 4; ++rr) {
        int row = arow + rr * 32;
        const float4 x4 = *(const float4*)&Af[(size_t)(m0 + row) * K + k0 + ak];
        As[row][slotA(ak + 0)] = f2bf(x4.x);
        As[row][slotA(ak + 1)] = f2bf(x4.y);
        As[row][slotA(ak + 2)] = f2bf(x4.z);
        As[row][slotA(ak + 3)] = f2bf(x4.w);
        if (k0 + 32 < K)
          __builtin_prefetch(&Af[(size_t)(m0 + row) * K + k0 + 32 + ak], 0, 3);
      }
    } else {
#pragma unroll
      for (int rr = 0; rr < 4; ++rr) {
        int row = arow + rr * 32;
        const uint2 u = *(const uint2*)&Ab[(size_t)(m0 + row) * K + k0 + ak];
        As[row][slotA(ak + 0)] = (unsigned short)(u.x & 0xffffu);
        As[row][slotA(ak + 1)] = (unsigned short)(u.x >> 16);
        As[row][slotA(ak + 2)] = (unsigned short)(u.y & 0xffffu);
        As[row][slotA(ak + 3)] = (unsigned short)(u.y >> 16);
        if (k0 + 32 < K)
          __builtin_prefetch(&Ab[(size_t)(m0 + row) * K + k0 + 32 + ak], 0, 3);
      }
    }
#pragma unroll
    for (int nn = 0; nn < 4; ++nn) {  // stage W transposed -> Bs[n][k]
      int n = bn0 + nn * 4;
      const float4 w4 = *(const float4*)&W[(size_t)(k0 + bk) * N + n0 + n];
      Bs[n + 0][bk] = f2bf(w4.x);
      Bs[n + 1][bk] = f2bf(w4.y);
      Bs[n + 2][bk] = f2bf(w4.z);
      Bs[n + 3][bk] = f2bf(w4.w);
    }
    if (k0 + 32 < K)
      __builtin_prefetch(&W[(size_t)(k0 + 32 + bk) * N + n0 + bn0], 0, 3);
    __syncthreads();
    v16bf a[2], b[4];
#pragma unroll
    for (int tm = 0; tm < 2; ++tm)
      a[tm] = frag_ld(&As[wm * 32 + tm * 16 + ll][lh * 16]);
#pragma unroll
    for (int tn = 0; tn < 4; ++tn)
      b[tn] = frag_ld(&Bs[wn * 64 + tn * 16 + ll][lh * 16]);
#pragma unroll
    for (int tm = 0; tm < 2; ++tm)
#pragma unroll
      for (int tn = 0; tn < 4; ++tn)
        acc[tm][tn] = wmma_bf16(a[tm], b[tn], acc[tm][tn]);
    __syncthreads();
  }

  // C/D layout: VGPR r -> row r (lanes 0-15) / row r+8 (lanes 16-31), col = ll.
#pragma unroll
  for (int tm = 0; tm < 2; ++tm)
#pragma unroll
    for (int tn = 0; tn < 4; ++tn)
#pragma unroll
      for (int r = 0; r < 8; ++r) {
        int row = m0 + wm * 32 + tm * 16 + r + lh * 8;
        int col = n0 + wn * 64 + tn * 16 + ll;
        float val = acc[tm][tn][r] + bias[col];
        if (MODE == 0) {
          const int Cd = N / 3;
          int s = col / Cd, c = col - s * Cd;
          unsigned short bv = f2bf(val);
          size_t idx = (size_t)row * Cd + c;
          if (s == 0)      oq[idx] = bv;
          else if (s == 1) ok[idx] = bv;
          else             ov[idx] = bv;
        } else {
          of[(size_t)row * N + col] = val;
        }
      }
}

// ---------------------------------------------------------------------------
// Per-head RMS norm on q and k: one wave per (b,l,h) row of 64 elements.
// ---------------------------------------------------------------------------
__global__ __launch_bounds__(256) void rms_kernel(
    unsigned short* __restrict__ q, unsigned short* __restrict__ k,
    const float* __restrict__ gq, const float* __restrict__ gk) {
  const int nrows = BB * LL * HH;
  int w = blockIdx.x * 8 + (threadIdx.x >> 5);
  int lane = threadIdx.x & 31;
  unsigned short* buf = q;
  const float* g = gq;
  int row = w;
  if (w >= nrows) { row = w - nrows; buf = k; g = gk; }
  size_t base = (size_t)row * DD + lane * 2;
  unsigned u = *(const unsigned*)&buf[base];
  float v0 = bf2f(u & 0xffffu), v1 = bf2f(u >> 16);
  float ss = v0 * v0 + v1 * v1;
#pragma unroll
  for (int m = 1; m < 32; m <<= 1) ss += __shfl_xor(ss, m, 32);
  float nrm = fmaxf(sqrtf(ss), 1e-12f);
  float sc = 8.0f / nrm;  // sqrt(D)/norm
  int h = row & (HH - 1);
  int d = lane * 2;
  v0 = v0 * g[h * DD + d] * sc;
  v1 = v1 * g[h * DD + d + 1] * sc;
  *(unsigned*)&buf[base] = (unsigned)f2bf(v0) | ((unsigned)f2bf(v1) << 16);
}

// ---------------------------------------------------------------------------
// Flash attention: block = one (b,h), 128 q rows; each wave owns 16 q rows.
// Per 32-key tile: 4 WMMAs for S = Q*K^T, online softmax, 4 WMMAs for O += P*V.
// K tile is staged with the CDNA5 async global->LDS path (ASYNCcnt), V tile
// goes through VGPRs because it is transposed on the way into LDS.
// ---------------------------------------------------------------------------
__global__ __launch_bounds__(256) void flash_kernel(
    const unsigned short* __restrict__ q, const unsigned short* __restrict__ k,
    const unsigned short* __restrict__ v, unsigned short* __restrict__ ho) {
  __shared__ __align__(16) unsigned short Ks[32][64];      // [key][d]
  __shared__ __align__(16) unsigned short Vt[64][32];      // [d][key]
  __shared__ __align__(16) unsigned short Ps[8][16][32];   // per-wave P relayout
  const int t = threadIdx.x, wid = t >> 5, lane = t & 31;
  const int lh = lane >> 4, ll = lane & 15;
  const int b = blockIdx.y >> 4, h = blockIdx.y & 15;
  const int q0 = blockIdx.x * 128 + wid * 16;

  // Q as two 16x32 A-fragments (d = 0..63), contiguous order (lane half h holds d = c*32 + 16h..16h+15)
  v16bf aq[2];
  {
    unsigned short tmp[16];
    size_t qb = ((size_t)(b * LL + q0 + ll)) * CC + h * DD;
#pragma unroll
    for (int c = 0; c < 2; ++c) {
#pragma unroll
      for (int i = 0; i < 16; ++i) {
        int d = c * 32 + lh * 16 + i;
        tmp[i] = q[qb + d];
      }
      __builtin_memcpy(&aq[c], tmp, 32);
    }
  }

  const v8f vz = {0.f, 0.f, 0.f, 0.f, 0.f, 0.f, 0.f, 0.f};
  v8f o[4] = {vz, vz, vz, vz};
  float mr[8], lr[8];
#pragma unroll
  for (int r = 0; r < 8; ++r) { mr[r] = -3.0e38f; lr[r] = 0.f; }

  const int skr = t & 31, sd = (t >> 5) * 8;  // staging: key row, d base
  // Per-lane constants for the async K-tile copy (GVS addressing):
  // lds byte offset of this lane's 16-byte chunk, and 32-bit global offset.
  const unsigned kLds = (unsigned)(size_t)&Ks[skr][sd];
  const unsigned kOff = (unsigned)((skr * CC + sd) * sizeof(unsigned short));

  for (int kt = 0; kt < LL; kt += 32) {
    // K tile: async global->LDS b128 copy, 256 lanes x 16B = 32x64 bf16 tile.
    {
      unsigned long long kBase =
          (unsigned long long)(const void*)(k + ((size_t)(b * LL + kt)) * CC +
                                            h * DD);
      asm volatile("global_load_async_to_lds_b128 %0, %1, %2"
                   :
                   : "v"(kLds), "v"(kOff), "s"(kBase)
                   : "memory");
    }
    size_t mb = ((size_t)(b * LL + kt + skr)) * CC + h * DD + sd;
    uint4 vd = *(const uint4*)&v[mb];
    const unsigned short* vs = (const unsigned short*)&vd;
#pragma unroll
    for (int j = 0; j < 8; ++j) Vt[sd + j][skr] = vs[j];  // transpose
    if (kt + 32 < LL) {  // prefetch next K/V tile into L2 (global_prefetch_b8)
      __builtin_prefetch(&k[mb + (size_t)32 * CC], 0, 3);
      __builtin_prefetch(&v[mb + (size_t)32 * CC], 0, 3);
    }
    asm volatile("s_wait_asynccnt 0x0" ::: "memory");  // K tile landed in LDS
    __syncthreads();

    // S (16 q x 32 keys) = two 16x16 tiles, K-dim = D = 64 chained
    v8f s0 = vz, s1 = vz;
#pragma unroll
    for (int c = 0; c < 2; ++c) {
      s0 = wmma_bf16(aq[c], frag_ld(&Ks[ll][c * 32 + lh * 16]), s0);
      s1 = wmma_bf16(aq[c], frag_ld(&Ks[16 + ll][c * 32 + lh * 16]), s1);
    }

    // Online softmax; each lane owns 8 rows (r -> row r + lh*8)
    float p0v[8], p1v[8], al[8];
#pragma unroll
    for (int r = 0; r < 8; ++r) {
      float x0 = s0[r] * 0.125f, x1 = s1[r] * 0.125f;  // 1/sqrt(D)
      float mx = fmaxf(x0, x1);
      mx = fmaxf(mx, __shfl_xor(mx, 1, 32));
      mx = fmaxf(mx, __shfl_xor(mx, 2, 32));
      mx = fmaxf(mx, __shfl_xor(mx, 4, 32));
      mx = fmaxf(mx, __shfl_xor(mx, 8, 32));
      float mn = fmaxf(mr[r], mx);
      al[r] = __expf(mr[r] - mn);
      p0v[r] = __expf(x0 - mn);
      p1v[r] = __expf(x1 - mn);
      float rs = p0v[r] + p1v[r];
      rs += __shfl_xor(rs, 1, 32);
      rs += __shfl_xor(rs, 2, 32);
      rs += __shfl_xor(rs, 4, 32);
      rs += __shfl_xor(rs, 8, 32);
      lr[r] = lr[r] * al[r] + rs;
      mr[r] = mn;
    }
#pragma unroll
    for (int dt = 0; dt < 4; ++dt)
#pragma unroll
      for (int r = 0; r < 8; ++r) o[dt][r] *= al[r];
#pragma unroll
    for (int r = 0; r < 8; ++r) {
      int pr = r + lh * 8;
      Ps[wid][pr][slotA(ll)]      = f2bf(p0v[r]);  // key col ll
      Ps[wid][pr][slotA(16 + ll)] = f2bf(p1v[r]);  // key col 16+ll
    }
    __syncthreads();  // make P relayout + Vt visible

    v16bf pa = frag_ld(&Ps[wid][ll][lh * 16]);  // P as 16x32 A-fragment
#pragma unroll
    for (int dt = 0; dt < 4; ++dt)
      o[dt] = wmma_bf16(pa, frag_ld(&Vt[dt * 16 + ll][lh * 16]), o[dt]);
    __syncthreads();
  }

#pragma unroll
  for (int r = 0; r < 8; ++r) {
    float inv = 1.0f / lr[r];
    int row = q0 + r + lh * 8;
    size_t base = ((size_t)(b * LL + row)) * CC + h * DD;
#pragma unroll
    for (int dt = 0; dt < 4; ++dt)
      ho[base + dt * 16 + ll] = f2bf(o[dt][r] * inv);
  }
}

// ---------------------------------------------------------------------------
extern "C" void kernel_launch(void* const* d_in, const int* in_sizes, int n_in,
                              void* d_out, int out_size, void* d_ws,
                              size_t ws_size, hipStream_t stream) {
  (void)in_sizes; (void)n_in; (void)out_size; (void)ws_size;
  const float* x    = (const float*)d_in[0];
  const float* Wqkv = (const float*)d_in[1];
  const float* bqkv = (const float*)d_in[2];
  const float* gq   = (const float*)d_in[3];
  const float* gk   = (const float*)d_in[4];
  const float* Wout = (const float*)d_in[5];
  const float* bout = (const float*)d_in[6];
  float* out = (float*)d_out;

  const size_t NE = (size_t)BB * LL * CC;  // 8,388,608 elems per buffer
  unsigned short* qws = (unsigned short*)d_ws;
  unsigned short* kws = qws + NE;
  unsigned short* vws = kws + NE;
  unsigned short* hws = vws + NE;

  dim3 blk(256);
  // 1) qkv = x @ Wqkv + b -> bf16 q/k/v
  gemm_bf16_wmma<0><<<dim3(3 * CC / 128, BB * LL / 128), blk, 0, stream>>>(
      x, nullptr, Wqkv, bqkv, qws, kws, vws, nullptr, BB * LL, 3 * CC, CC);
  // 2) per-head RMS norm on q and k (2 * B*L*H rows, 8 waves/block)
  rms_kernel<<<dim3(2 * BB * LL * HH / 8), blk, 0, stream>>>(qws, kws, gq, gk);
  // 3) flash attention -> bf16 h
  flash_kernel<<<dim3(LL / 128, BB * HH), blk, 0, stream>>>(qws, kws, vws, hws);
  // 4) out = h @ Wout + bout (f32)
  gemm_bf16_wmma<1><<<dim3(CC / 128, BB * LL / 128), blk, 0, stream>>>(
      nullptr, hws, Wout, bout, nullptr, nullptr, nullptr, out, BB * LL, CC, CC);
}
